// DorefaConv2d_27522150433246
// MI455X (gfx1250) — 8-bit matrix operand order repaired and checked by reading, compiled, not run on hardware
//
#include <hip/hip_runtime.h>
#include <stdint.h>

typedef int v8i __attribute__((ext_vector_type(8)));
typedef int vec4i __attribute__((vector_size(16)));

#define NB    32
#define HH    56
#define WW    56
#define CIN   256
#define COUT  256
#define NKC   36
#define WELEMS (3*3*256*256)
#define XELEMS (NB*HH*WW*CIN)

#define WS_ACC_OFF   0
#define WS_BPACK_OFF 1024
#define WS_XQ_OFF    (1u << 20)

#define XSLOTS 72
#define ROWB   (XSLOTS*CIN)
#define SMEMB  (3*ROWB)

#if defined(__has_builtin)
#  if __has_builtin(__builtin_amdgcn_global_load_async_to_lds_b128)
#    define HAVE_ASYNC_LDS 1
#  endif
#  if __has_builtin(__builtin_amdgcn_s_wait_asynccnt)
#    define HAVE_WAIT_ASYNC 1
#  endif
#endif

__global__ void k_init(float* acc) {
  if (threadIdx.x == 0) acc[0] = 0.0f;
}

__global__ void k_abssum(const float* __restrict__ W, float* __restrict__ acc) {
  __shared__ float red[256];
  float s = 0.0f;
  for (int i = blockIdx.x * blockDim.x + threadIdx.x; i < WELEMS;
       i += gridDim.x * blockDim.x)
    s += fabsf(W[i]);
  red[threadIdx.x] = s;
  __syncthreads();
  for (int o = 128; o > 0; o >>= 1) {
    if ((int)threadIdx.x < o) red[threadIdx.x] += red[threadIdx.x + o];
    __syncthreads();
  }
  if (threadIdx.x == 0) atomicAdd(acc, red[0]);
}

__global__ void k_quant_act(const float* __restrict__ x,
                            unsigned char* __restrict__ xq) {
  int i = blockIdx.x * blockDim.x + threadIdx.x;
  int base = i * 4;
  if (base >= XELEMS) return;
  float4 v = *(const float4*)(x + base);
  unsigned q0 = (unsigned)rintf(fminf(1.0f, fabsf(v.x)) * 7.0f);
  unsigned q1 = (unsigned)rintf(fminf(1.0f, fabsf(v.y)) * 7.0f);
  unsigned q2 = (unsigned)rintf(fminf(1.0f, fabsf(v.z)) * 7.0f);
  unsigned q3 = (unsigned)rintf(fminf(1.0f, fabsf(v.w)) * 7.0f);
  *(unsigned*)(xq + base) = q0 | (q1 << 8) | (q2 << 16) | (q3 << 24);
}

__global__ void k_pack_w(const float* __restrict__ W, signed char* __restrict__ bp) {
  int i = blockIdx.x * blockDim.x + threadIdx.x;
  if (i >= WELEMS) return;
  int t    = i / 36864;
  int r    = i - t * 36864;
  int kc   = r >> 10;
  int j    = r & 1023;
  int lane = j >> 5;
  int v    = (j >> 2) & 7;
  int b    = j & 3;
  int klocal = ((v >> 1) << 4) + ((lane >> 4) << 3) + ((v & 1) << 2) + b;
  int k    = kc * 64 + klocal;
  int tap  = k >> 8;
  int c    = k & 255;
  int cout = (t << 4) + (lane & 15);
  float w  = W[(tap * 256 + c) * 256 + cout];
  bp[i] = (w > 0.0f) ? (signed char)1 : ((w < 0.0f) ? (signed char)-1 : (signed char)0);
}

__global__ __launch_bounds__(256)
void k_conv(const unsigned char* __restrict__ xq,
            const signed char*  __restrict__ bp,
            const float*        __restrict__ acc,
            const float*        __restrict__ bias,
            float*              __restrict__ out) {
  __shared__ __align__(16) unsigned char smem[SMEMB];
  const int tid = threadIdx.x;
  const int bid = blockIdx.x;
  const int n = bid / HH;
  const int h = bid % HH;

  for (int i = tid; i < SMEMB / 16; i += 256)
    ((uint4*)smem)[i] = make_uint4(0u, 0u, 0u, 0u);
  __syncthreads();

  for (int i = tid; i < 3 * WW * 16; i += 256) {
    int kh = i / (WW * 16);
    int rr = i - kh * (WW * 16);
    int x  = rr >> 4;
    int q  = rr & 15;
    int hin = h + kh - 1;
    if ((unsigned)hin < (unsigned)HH) {
      const unsigned char* src =
          xq + (size_t)(((n * HH + hin) * WW + x) * CIN) + q * 16;
      unsigned char* dst = smem + kh * ROWB + (x + 1) * CIN + q * 16;
#if defined(HAVE_ASYNC_LDS)
      __builtin_amdgcn_global_load_async_to_lds_b128(
          (__attribute__((address_space(1))) vec4i*)(void*)src,
          (__attribute__((address_space(3))) vec4i*)(void*)dst,
          0, 0);
#else
      *(uint4*)dst = *(const uint4*)src;
#endif
    }
  }
#if defined(HAVE_ASYNC_LDS)
#  if defined(HAVE_WAIT_ASYNC)
  __builtin_amdgcn_s_wait_asynccnt(0);
#  else
  asm volatile("s_wait_asynccnt 0x0" ::: "memory");
#  endif
#endif
  __syncthreads();

  const float E     = acc[0] * (1.0f / 589824.0f);
  const float scale = E * (1.0f / 7.0f);

  const int lane   = tid & 31;
  const int wid    = tid >> 5;
  const int m_base = (wid & 3) << 4;
  const int n_half = wid >> 2;
  const int mrow   = m_base + (lane & 15);
  const int khalf8 = (lane >> 4) << 3;

  v8i accv[8];
  #pragma unroll
  for (int t8 = 0; t8 < 8; ++t8)
    accv[t8] = (v8i){0, 0, 0, 0, 0, 0, 0, 0};

  const signed char* btile = bp + (size_t)(n_half * 8) * 36864 + lane * 32;

  for (int kc = 0; kc < NKC; ++kc) {
    const int tap = kc >> 2;
    const int cc  = kc & 3;
    const int kh  = tap / 3;
    const int kw  = tap - kh * 3;

    const unsigned abase =
        (unsigned)(kh * ROWB + (mrow + kw) * CIN + cc * 64 + khalf8);
    union { unsigned long long q[4]; v8i v; } au;
    #pragma unroll
    for (int q = 0; q < 4; ++q)
      au.q[q] = *(const unsigned long long*)(smem + abase + q * 16);
    const v8i afrag = au.v;

    const signed char* bbase = btile + (size_t)kc * 1024;
    if (kc + 1 < NKC) __builtin_prefetch(bbase + 1024, 0, 0);

    v8i bfr[8];
    #pragma unroll
    for (int t8 = 0; t8 < 8; ++t8)
      bfr[t8] = *(const v8i*)(bbase + (size_t)t8 * 36864);

    #pragma unroll
    for (int t8 = 0; t8 < 8; ++t8) {
      accv[t8] = __builtin_amdgcn_wmma_i32_16x16x64_iu8(
           false, afrag,  true, bfr[t8], accv[t8],
           false,  false);
    }
  }

  #pragma unroll
  for (int t8 = 0; t8 < 8; ++t8) {
    const int cout = ((n_half * 8 + t8) << 4) + (lane & 15);
    const float bv = bias[cout];
    #pragma unroll
    for (int r = 0; r < 8; ++r) {
      const int w_out = m_base + r + ((lane >> 4) << 3);
      if (w_out < WW) {
        out[(size_t)((n * HH + h) * WW + w_out) * COUT + cout] =
            (float)accv[t8][r] * scale + bv;
      }
    }
  }
}

extern "C" void kernel_launch(void* const* d_in, const int* in_sizes, int n_in,
                              void* d_out, int out_size, void* d_ws, size_t ws_size,
                              hipStream_t stream) {
  (void)in_sizes; (void)n_in; (void)out_size; (void)ws_size;
  const float* x = (const float*)d_in[0];
  const float* W = (const float*)d_in[1];
  const float* b = (const float*)d_in[2];
  float* out = (float*)d_out;

  char* ws = (char*)d_ws;
  float*         accum = (float*)(ws + WS_ACC_OFF);
  signed char*   bpack = (signed char*)(ws + WS_BPACK_OFF);
  unsigned char* xqb   = (unsigned char*)(ws + WS_XQ_OFF);

  k_init<<<1, 64, 0, stream>>>(accum);
  k_abssum<<<512, 256, 0, stream>>>(W, accum);
  k_quant_act<<<(XELEMS / 4 + 255) / 256, 256, 0, stream>>>(x, xqb);
  k_pack_w<<<(WELEMS + 255) / 256, 256, 0, stream>>>(W, bpack);
  k_conv<<<NB * HH, 256, 0, stream>>>(xqb, bpack, accum, b, out);
}
